// SingleResoDictPlenoxels_90314572301120
// MI455X (gfx1250) — matrix operand order repaired and checked by reading, not compiled, not run on hardware
//
#include <hip/hip_runtime.h>
#include <hip/hip_bf16.h>

typedef __attribute__((ext_vector_type(16))) _Float16 v16h;
typedef __attribute__((ext_vector_type(8)))  float    v8f;

#define RADIUS  1.3f
#define STEPF   0.020634920634920634f   /* RADIUS / 63 */
#define NSAMP   217
#define NTILE   14                      /* 14*16 = 224 >= 217 */
#define RGBBASE 0
#define DBASE   3072                    /* 1024*3 */
#define ABASE   4096                    /* + 1024 */
#define SCALIDX 226304                  /* 4096 + 1024*217 */

__device__ __forceinline__ float bpermf(int srcLane, float v) {
  return __int_as_float(__builtin_amdgcn_ds_bpermute(srcLane * 4, __float_as_int(v)));
}
__device__ __forceinline__ int bpermi(int srcLane, int v) {
  return __builtin_amdgcn_ds_bpermute(srcLane * 4, v);
}

__global__ __launch_bounds__(128, 1) void plen_kernel(
    const float* __restrict__ rays_o, const float* __restrict__ rays_d,
    const float* __restrict__ cgrid,  const float* __restrict__ atoms,
    float* __restrict__ out)
{
  // Pre-swizzled WMMA A-fragments of atoms^T: [mtile(2)][kstep(32)][lane(32)][h(16)] fp16 = 64 KB
  __shared__ _Float16 aFrag[32768];

  const int tid = threadIdx.x;

  // ---- one-time LDS fill: A(m=dim, k=cell*16+atom) in fragment order ----
  #pragma unroll 4
  for (int i = 0; i < 256; ++i) {
    int E    = tid + i * 128;
    int h    = E & 15;
    int ln   = (E >> 4) & 31;
    int s    = (E >> 9) & 31;
    int mt   = E >> 14;
    // A fragment, contiguous k order shared with B: lanes<16 hold K {0..15}, lanes>=16 hold K {16..31}
    int kloc = h + ((ln & 16) ? 16 : 0);
    int K    = s * 32 + kloc;
    int M    = (ln & 15) + mt * 16;
    float v  = 0.f;
    if (M < 28) v = atoms[((K >> 4) * 16 + (K & 15)) * 28 + M];
    aFrag[E] = (_Float16)v;
  }
  __syncthreads();

  const int  lane   = tid & 31;
  const int  wave   = tid >> 5;
  const int  ray    = blockIdx.x * 4 + wave;     // grid = 256 blocks -> 1024 rays
  const int  sub    = lane & 15;
  const bool hiHalf = (lane & 16) != 0;

  // ---- per-ray setup (uniform across wave) ----
  float ox = rays_o[ray*3+0], oy = rays_o[ray*3+1], oz = rays_o[ray*3+2];
  float dx = rays_d[ray*3+0], dy = rays_d[ray*3+1], dz = rays_d[ray*3+2];

  float sdx = (fabsf(dx) < 1e-9f) ? 1e-9f : dx;
  float sdy = (fabsf(dy) < 1e-9f) ? 1e-9f : dy;
  float sdz = (fabsf(dz) < 1e-9f) ? 1e-9f : dz;
  float t1x = (-RADIUS - ox) / sdx, t2x = (RADIUS - ox) / sdx;
  float t1y = (-RADIUS - oy) / sdy, t2y = (RADIUS - oy) / sdy;
  float t1z = (-RADIUS - oz) / sdz, t2z = (RADIUS - oz) / sdz;
  float tnear = fmaxf(fmaxf(fmaxf(fminf(t1x,t2x), fminf(t1y,t2y)), fminf(t1z,t2z)), 0.f);
  float tfar  = fminf(fminf(fmaxf(t1x,t2x), fmaxf(t1y,t2y)), fmaxf(t1z,t2z));

  float dnorm = sqrtf(dx*dx + dy*dy + dz*dz);
  float delta = STEPF * dnorm;
  float inl   = 1.f / dnorm;
  float ux = dx * inl, uy = dy * inl, uz = dz * inl;

  float sh[9];
  sh[0] =  0.28209479177387814f;
  sh[1] = -0.4886025119029199f * uy;
  sh[2] =  0.4886025119029199f * uz;
  sh[3] = -0.4886025119029199f * ux;
  sh[4] =  1.0925484305920792f * ux * uy;
  sh[5] = -1.0925484305920792f * uy * uz;
  sh[6] =  0.31539156525252005f * (2.f*uz*uz - ux*ux - uy*uy);
  sh[7] = -1.0925484305920792f * ux * uz;
  sh[8] =  0.5462742152960396f * (ux*ux - uy*uy);

  float Tc = 1.f;
  float rAcc0 = 0.f, rAcc1 = 0.f, rAcc2 = 0.f, dAcc = 0.f, wAcc = 0.f;

  #pragma unroll 1
  for (int t = 0; t < NTILE; ++t) {
    const int  sidx = t * 16 + sub;          // both half-waves mirror sample sub
    const float tm  = tnear + ((float)sidx + 0.5f) * STEPF;
    const bool geom = (tm < tfar) && (tfar > tnear);

    float px = ox + dx * tm, py = oy + dy * tm, pz = oz + dz * tm;
    float p01x = fminf(fmaxf((px + RADIUS) * (0.5f / RADIUS), 0.f), 1.f - 1e-6f);
    float p01y = fminf(fmaxf((py + RADIUS) * (0.5f / RADIUS), 0.f), 1.f - 1e-6f);
    float p01z = fminf(fmaxf((pz + RADIUS) * (0.5f / RADIUS), 0.f), 1.f - 1e-6f);

    // ---- coarse trilinear -> per-point atom coefficients (fp32) ----
    float ccx = p01x * 15.f, ccy = p01y * 15.f, ccz = p01z * 15.f;
    float flx = floorf(ccx), fly = floorf(ccy), flz = floorf(ccz);
    float fcx = ccx - flx, fcy = ccy - fly, fcz = ccz - flz;
    int i0x = min(max((int)flx, 0), 15), i1x = min(i0x + 1, 15);
    int i0y = min(max((int)fly, 0), 15), i1y = min(i0y + 1, 15);
    int i0z = min(max((int)flz, 0), 15), i1z = min(i0z + 1, 15);

    float cacc[16];
    #pragma unroll
    for (int a = 0; a < 16; ++a) cacc[a] = 0.f;
    #pragma unroll
    for (int c = 0; c < 8; ++c) {
      int ix = (c & 4) ? i1x : i0x;
      int iy = (c & 2) ? i1y : i0y;
      int iz = (c & 1) ? i1z : i0z;
      float w = ((c & 4) ? fcx : 1.f - fcx) *
                ((c & 2) ? fcy : 1.f - fcy) *
                ((c & 1) ? fcz : 1.f - fcz);
      const float4* row = (const float4*)(cgrid + (((ix * 16 + iy) * 16 + iz) << 4));
      float4 r0 = row[0], r1 = row[1], r2 = row[2], r3 = row[3];
      cacc[0]  += w * r0.x; cacc[1]  += w * r0.y; cacc[2]  += w * r0.z; cacc[3]  += w * r0.w;
      cacc[4]  += w * r1.x; cacc[5]  += w * r1.y; cacc[6]  += w * r1.z; cacc[7]  += w * r1.w;
      cacc[8]  += w * r2.x; cacc[9]  += w * r2.y; cacc[10] += w * r2.z; cacc[11] += w * r2.w;
      cacc[12] += w * r3.x; cacc[13] += w * r3.y; cacc[14] += w * r3.z; cacc[15] += w * r3.w;
    }
    v16h cfrag;
    #pragma unroll
    for (int a = 0; a < 16; ++a) cfrag[a] = (_Float16)cacc[a];

    // ---- fine-local cell / weights ----
    float pfx = p01x * 16.f, pfy = p01y * 16.f, pfz = p01z * 16.f;
    float lfx = (pfx - floorf(pfx)) * 3.f;
    float lfy = (pfy - floorf(pfy)) * 3.f;
    float lfz = (pfz - floorf(pfz)) * 3.f;
    float j0fx = floorf(lfx), j0fy = floorf(lfy), j0fz = floorf(lfz);
    float ffx = lfx - j0fx, ffy = lfy - j0fy, ffz = lfz - j0fz;
    int j0x = min(max((int)j0fx, 0), 3);
    int j0y = min(max((int)j0fy, 0), 3);
    int j0z = min(max((int)j0fz, 0), 3);

    // occupancy mask of the 8 corner cells (cell = jx*16+jy*4+jz)
    unsigned long long cm = 0ull;
    #pragma unroll
    for (int c = 0; c < 8; ++c) {
      int bit = (j0x + ((c >> 2) & 1)) * 16 + (j0y + ((c >> 1) & 1)) * 4 + (j0z + (c & 1));
      cm |= (1ull << bit);
    }
    unsigned lo = (unsigned)cm, hi = (unsigned)(cm >> 32);
    #pragma unroll
    for (int k = 1; k < 32; k <<= 1) {
      lo |= (unsigned)bpermi(lane ^ k, (int)lo);
      hi |= (unsigned)bpermi(lane ^ k, (int)hi);
    }
    unsigned smlo = (unsigned)__builtin_amdgcn_readfirstlane((int)lo);
    unsigned smhi = (unsigned)__builtin_amdgcn_readfirstlane((int)hi);

    // ---- WMMA: outT(dim, point) += atomsT(dim, K) * wcoeff(K, point) ----
    v8f acc0 = {}; v8f acc1 = {};
    #pragma unroll 1
    for (int s = 0; s < 32; ++s) {
      unsigned pair = (s < 16) ? ((smlo >> (2 * s)) & 3u) : ((smhi >> (2 * s - 32)) & 3u);
      if (pair == 0u) continue;                        // scalar branch: EXEC stays all-ones
      // issue both A-fragment loads first so DS latency hides under the weight math
      v16h a0 = *(const v16h*)(&aFrag[(s * 32 + lane) * 16]);
      v16h a1 = *(const v16h*)(&aFrag[16384 + (s * 32 + lane) * 16]);
      int cell = 2 * s + (hiHalf ? 1 : 0);             // B half-lane <-> one 16-atom cell
      int gx = cell >> 4, gy = (cell >> 2) & 3, gz = cell & 3;
      float wx = (gx == j0x) ? (1.f - ffx) : ((gx == j0x + 1) ? ffx : 0.f);
      float wy = (gy == j0y) ? (1.f - ffy) : ((gy == j0y + 1) ? ffy : 0.f);
      float wz = (gz == j0z) ? (1.f - ffz) : ((gz == j0z + 1) ? ffz : 0.f);
      float scale = wx * wy * wz;
      v16h b = cfrag * (_Float16)scale;                // 8x v_pk_mul_f16, in-register B fragment
      acc0 = __builtin_amdgcn_wmma_f32_16x16x32_f16(false, a0, false, b, (short)0, acc0, false, false);
      acc1 = __builtin_amdgcn_wmma_f32_16x16x32_f16(false, a1, false, b, (short)0, acc1, false, false);
    }

    // ---- SH color + sigma from D-tile layout (M=dim, N=point) ----
    float p0, p1, p2, sg;
    if (!hiHalf) {                    // this lane holds dims 0..7 (acc0) and 16..23 (acc1)
      p0 = acc0[0]*sh[0] + acc0[1]*sh[1] + acc0[2]*sh[2] + acc0[3]*sh[3] +
           acc0[4]*sh[4] + acc0[5]*sh[5] + acc0[6]*sh[6] + acc0[7]*sh[7];
      p1 = acc1[0]*sh[7] + acc1[1]*sh[8];
      p2 = acc1[2]*sh[0] + acc1[3]*sh[1] + acc1[4]*sh[2] +
           acc1[5]*sh[3] + acc1[6]*sh[4] + acc1[7]*sh[5];
      sg = 0.f;
    } else {                          // dims 8..15 (acc0) and 24..31 (acc1); dim 27 = sigma
      p0 = acc0[0]*sh[8];
      p1 = acc0[1]*sh[0] + acc0[2]*sh[1] + acc0[3]*sh[2] + acc0[4]*sh[3] +
           acc0[5]*sh[4] + acc0[6]*sh[5] + acc0[7]*sh[6];
      p2 = acc1[0]*sh[6] + acc1[1]*sh[7] + acc1[2]*sh[8];
      sg = acc1[3];
    }
    float q0 = p0 + bpermf(lane ^ 16, p0);
    float q1 = p1 + bpermf(lane ^ 16, p1);
    float q2 = p2 + bpermf(lane ^ 16, p2);
    float sigma = sg + bpermf(lane ^ 16, sg);          // lanes<16: sigma of their point

    // ---- compositing: prefix product over 16 samples (lanes 0..15) ----
    const bool inr  = (!hiHalf) && (sidx < NSAMP);
    float sigm  = (inr && geom) ? fmaxf(sigma, 0.f) : 0.f;
    float alpha = 1.f - expf(-sigm * delta);
    float fct   = inr ? (1.f - alpha + 1e-10f) : 1.f;
    float incl  = fct;
    #pragma unroll
    for (int k = 1; k <= 8; k <<= 1) {
      float oth = bpermf(lane - k, incl);
      incl *= (lane >= k) ? oth : 1.f;
    }
    float excl  = bpermf(lane - 1, incl);
    excl = (lane == 0) ? 1.f : excl;
    float trans = Tc * excl;
    float wgt   = alpha * trans;                       // zero on lanes>=16 and invalid samples
    rAcc0 += wgt / (1.f + expf(-q0));
    rAcc1 += wgt / (1.f + expf(-q1));
    rAcc2 += wgt / (1.f + expf(-q2));
    dAcc  += wgt * tm;
    wAcc  += wgt;
    // carry = product over this tile (broadcast lane 15 via readlane -> scalar mul)
    Tc *= __int_as_float(__builtin_amdgcn_readlane(__float_as_int(incl), 15));
    if (inr) out[ABASE + ray * NSAMP + sidx] = alpha;
  }

  // ---- wave reduction + final writes ----
  #pragma unroll
  for (int k = 16; k >= 1; k >>= 1) {
    rAcc0 += bpermf(lane ^ k, rAcc0);
    rAcc1 += bpermf(lane ^ k, rAcc1);
    rAcc2 += bpermf(lane ^ k, rAcc2);
    dAcc  += bpermf(lane ^ k, dAcc);
    wAcc  += bpermf(lane ^ k, wAcc);
  }
  if (lane == 0) {
    float bg = 1.f - wAcc;                             // white background
    out[RGBBASE + ray * 3 + 0] = rAcc0 + bg;
    out[RGBBASE + ray * 3 + 1] = rAcc1 + bg;
    out[RGBBASE + ray * 3 + 2] = rAcc2 + bg;
    out[DBASE + ray] = dAcc;
  }
  if (ray == 0 && tid == 0) out[SCALIDX] = 0.f;        // trailing scalar output
}

extern "C" void kernel_launch(void* const* d_in, const int* in_sizes, int n_in,
                              void* d_out, int out_size, void* d_ws, size_t ws_size,
                              hipStream_t stream) {
  const float* rays_o = (const float*)d_in[0];
  const float* rays_d = (const float*)d_in[1];
  const float* cgrid  = (const float*)d_in[2];
  const float* atoms  = (const float*)d_in[3];
  float* out = (float*)d_out;
  // 1024 rays, 1 wave32 per ray, 4 waves per block -> 256 blocks
  plen_kernel<<<256, 128, 0, stream>>>(rays_o, rays_d, cgrid, atoms, out);
}
